// CrossAttention_7782480740849
// MI455X (gfx1250) — matrix operand order repaired and checked by reading, not compiled, not run on hardware
//
#include <hip/hip_runtime.h>

// ---------- types ----------
typedef __attribute__((ext_vector_type(16))) __bf16          v16bf;
typedef __attribute__((ext_vector_type(8)))  float           v8f;
typedef __attribute__((ext_vector_type(8)))  unsigned short  us8;

union Frag16 {
    struct { us8 lo, hi; } p;
    v16bf v;
};

__device__ __forceinline__ unsigned short f32_to_bf16(float f) {
    union { float f; unsigned int u; } v; v.f = f;
    unsigned int u = v.u;
    unsigned int r = u + 0x7FFFu + ((u >> 16) & 1u);   // round-to-nearest-even
    return (unsigned short)(r >> 16);
}

__device__ __forceinline__ float wave_sum(float v) {
    #pragma unroll
    for (int off = 16; off; off >>= 1) v += __shfl_xor(v, off, 32);
    return v;
}
__device__ __forceinline__ float wave_max(float v) {
    #pragma unroll
    for (int off = 16; off; off >>= 1) v = fmaxf(v, __shfl_xor(v, off, 32));
    return v;
}

// ---------- weight pack: f32 [K,N] row-major -> bf16 WMMA-fragment-major, K/N padded ----------
// Block (kb,nb): K rows [kb*32,kb*32+32) (kb < KBp, zero for k >= K),
//                N cols [nb*16,nb*16+16) (nb < NBp, zero for n >= N).
// Block storage: 512 halves at ((nb*KBp + kb)*512); lane (c + 16*s) owns halves
// [lane*16 .. lane*16+15] = W[kb*32 + 16*s + i][nb*16 + c], i = 0..15.
__global__ void __launch_bounds__(256)
pack_weight_bf16(const float* __restrict__ W, unsigned short* __restrict__ out,
                 int K, int N, int KBp, int NBp) {
    long total = (long)KBp * NBp * 512;
    long t = (long)blockIdx.x * 256 + threadIdx.x;
    if (t >= total) return;
    int i    = (int)(t & 15);
    int lane = (int)((t >> 4) & 31);
    long blk = t >> 9;
    int kb = (int)(blk % KBp);
    int nb = (int)(blk / KBp);
    int k = kb * 32 + (lane >> 4) * 16 + i;
    int n = nb * 16 + (lane & 15);
    float v = (k < K && n < N) ? W[(long)k * N + n] : 0.0f;
    out[t] = f32_to_bf16(v);
}

// ---------- LayerNorm over C=768, optional residual add, f32 + bf16 outputs ----------
__global__ void __launch_bounds__(256)
ln_kernel(const float* __restrict__ in, const float* __restrict__ add,
          const float* __restrict__ g, const float* __restrict__ bta,
          float* __restrict__ out_f32, unsigned short* __restrict__ out_bf16) {
    const int C = 768;
    long row = blockIdx.x;
    int  t   = threadIdx.x;            // 256 threads, 3 elems each
    const float* p = in + row * C;

    float vals[3];
    float s = 0.f, ss = 0.f;
    #pragma unroll
    for (int j = 0; j < 3; ++j) {
        int i = t + j * 256;
        float v = p[i];
        if (add) v += add[row * C + i];
        vals[j] = v; s += v; ss += v * v;
    }
    float ws1 = wave_sum(s);
    float ws2 = wave_sum(ss);
    __shared__ float sm[16];
    int wave = t >> 5, lane = t & 31;
    if (lane == 0) { sm[wave] = ws1; sm[8 + wave] = ws2; }
    __syncthreads();
    float tot = 0.f, tot2 = 0.f;
    #pragma unroll
    for (int w = 0; w < 8; ++w) { tot += sm[w]; tot2 += sm[8 + w]; }
    float mu  = tot * (1.0f / 768.0f);
    float var = tot2 * (1.0f / 768.0f) - mu * mu;
    float rs  = rsqrtf(var + 1e-6f);
    #pragma unroll
    for (int j = 0; j < 3; ++j) {
        int i = t + j * 256;
        float o = (vals[j] - mu) * rs * g[i] + bta[i];
        if (out_f32)  out_f32[row * C + i]  = o;
        if (out_bf16) out_bf16[row * C + i] = f32_to_bf16(o);
    }
}

// ---------- elementwise attention: softmax(q*k*scale) * v per (combo,b,head) ----------
// qkv: [2B, 2304] f32 (rows 0..B-1 = x, B..2B-1 = y; cols: q|k|v each 768, head-major)
// gout: [4B*8, 128] bf16 (per (combo,b,head) row: 96 real values + 32 zero pad)
__global__ void __launch_bounds__(256)
attn_kernel(const float* __restrict__ qkv, unsigned short* __restrict__ gout, int B) {
    long blk = blockIdx.x;             // 4*B blocks
    int  c   = (int)(blk / B);
    long b   = blk % B;
    int  h    = threadIdx.x >> 5;      // 8 waves = 8 heads
    int  lane = threadIdx.x & 31;

    long qrow, krow, vrow;
    switch (c) {
        case 0:  qrow = b;     krow = b;     vrow = b;     break;  // x,x,x
        case 1:  qrow = B + b; krow = B + b; vrow = B + b; break;  // y,y,y
        case 2:  qrow = B + b; krow = b;     vrow = b;     break;  // q_y,k_x,v_x
        default: qrow = b;     krow = B + b; vrow = B + b; break;  // q_x,k_y,v_y
    }
    const float scale = rsqrtf(96.0f);           // hd^-0.5
    long qb = qrow * 2304 +        h * 96;
    long kb = krow * 2304 +  768 + h * 96;
    long vb = vrow * 2304 + 1536 + h * 96;

    float tv[3], vv[3];
    float m = -1e30f;
    #pragma unroll
    for (int j = 0; j < 3; ++j) {
        int d = lane + j * 32;
        float t = qkv[qb + d] * qkv[kb + d] * scale;
        vv[j] = qkv[vb + d];
        tv[j] = t;
        m = fmaxf(m, t);
    }
    m = wave_max(m);
    float s = 0.f;
    #pragma unroll
    for (int j = 0; j < 3; ++j) { float e = __expf(tv[j] - m); tv[j] = e; s += e; }
    s = wave_sum(s);
    float inv = 1.0f / s;
    long ob = blk * 1024 + h * 128;    // K padded 96 -> 128
    #pragma unroll
    for (int j = 0; j < 3; ++j) {
        int d = lane + j * 32;
        gout[ob + d] = f32_to_bf16(tv[j] * inv * vv[j]);
    }
    gout[ob + 96 + lane] = 0;          // zero the K pad
}

// ---------- generic bf16 WMMA GEMM: out[M,N] = A[M,K] @ Bpack + bias + resid ----------
// A: bf16 row-major (K stride). Bpack: fragment-major, cols padded to mult of 64,
// K-blocks = K/32 and MUST be even. fp32 accumulate. Wave computes a 16x64 tile.
// Ping-pong software pipeline: fragment set for step k+1 loads during WMMAs of step k.
// Only the epilogue guards col < N. 8 waves per block -> 128-row block tile.
__global__ void __launch_bounds__(256)
gemm_bf16(const unsigned short* __restrict__ A, const unsigned short* __restrict__ Bp,
          const float* __restrict__ bias, const float* __restrict__ resid,
          float* __restrict__ out, int M, int N, int K) {
    const int KB = K >> 5;                    // even by construction
    int wave = threadIdx.x >> 5;
    int lane = threadIdx.x & 31;
    int m0 = (blockIdx.x * 8 + wave) * 16;
    int n0 = blockIdx.y * 64;
    if (m0 >= M) return;                      // wave-uniform
    int nb0 = n0 >> 4;

    v8f acc0 = {}, acc1 = {}, acc2 = {}, acc3 = {};

    int arow = m0 + (lane & 15);
    int koff = (lane >> 4) * 16;
    const unsigned short* Ap = A + (long)arow * K + koff;
    const unsigned short* Bq = Bp + (long)nb0 * KB * 512 + lane * 16;
    const long bs = (long)KB * 512;           // halves between adjacent N-blocks

    Frag16 aX, b0X, b1X, b2X, b3X;            // set X
    Frag16 aY, b0Y, b1Y, b2Y, b3Y;            // set Y

    #define LOADSET(a_, b0_, b1_, b2_, b3_, ap_, bq_)                          \
        a_.p.lo  = *(const us8*)(ap_);        a_.p.hi  = *(const us8*)((ap_) + 8); \
        b0_.p.lo = *(const us8*)(bq_);        b0_.p.hi = *(const us8*)((bq_) + 8);  \
        b1_.p.lo = *(const us8*)((bq_) + bs);     b1_.p.hi = *(const us8*)((bq_) + bs + 8);     \
        b2_.p.lo = *(const us8*)((bq_) + 2 * bs); b2_.p.hi = *(const us8*)((bq_) + 2 * bs + 8); \
        b3_.p.lo = *(const us8*)((bq_) + 3 * bs); b3_.p.hi = *(const us8*)((bq_) + 3 * bs + 8)

    #define WMMASET(a_, b0_, b1_, b2_, b3_)                                                          \
        acc0 = __builtin_amdgcn_wmma_f32_16x16x32_bf16(false, a_.v, false, b0_.v, (short)0, acc0, false, false); \
        acc1 = __builtin_amdgcn_wmma_f32_16x16x32_bf16(false, a_.v, false, b1_.v, (short)0, acc1, false, false); \
        acc2 = __builtin_amdgcn_wmma_f32_16x16x32_bf16(false, a_.v, false, b2_.v, (short)0, acc2, false, false); \
        acc3 = __builtin_amdgcn_wmma_f32_16x16x32_bf16(false, a_.v, false, b3_.v, (short)0, acc3, false, false)

    LOADSET(aX, b0X, b1X, b2X, b3X, Ap, Bq);               // set X <- block 0
    for (int kb = 0; kb < KB - 2; kb += 2) {
        LOADSET(aY, b0Y, b1Y, b2Y, b3Y, Ap + 32, Bq + 512); // set Y <- kb+1
        __builtin_prefetch(Ap + 96, 0, 1);
        WMMASET(aX, b0X, b1X, b2X, b3X);                    // compute kb
        Ap += 64; Bq += 1024;
        LOADSET(aX, b0X, b1X, b2X, b3X, Ap, Bq);            // set X <- kb+2
        WMMASET(aY, b0Y, b1Y, b2Y, b3Y);                    // compute kb+1
    }
    LOADSET(aY, b0Y, b1Y, b2Y, b3Y, Ap + 32, Bq + 512);     // set Y <- KB-1
    WMMASET(aX, b0X, b1X, b2X, b3X);                        // compute KB-2
    WMMASET(aY, b0Y, b1Y, b2Y, b3Y);                        // compute KB-1

    #undef LOADSET
    #undef WMMASET

    int r0 = m0 + 8 * (lane >> 4);
    int c0 = n0 + (lane & 15);
    auto store_tile = [&](const v8f& acc, int j) {
        int col = c0 + j * 16;
        if (col < N) {
            float badd = bias ? bias[col] : 0.0f;
            #pragma unroll
            for (int i = 0; i < 8; ++i) {
                long idx = (long)(r0 + i) * N + col;
                float v = acc[i] + badd;
                if (resid) v += resid[idx];
                out[idx] = v;
            }
        }
    };
    store_tile(acc0, 0);
    store_tile(acc1, 1);
    store_tile(acc2, 2);
    store_tile(acc3, 3);
}

// ---------- host ----------
extern "C" void kernel_launch(void* const* d_in, const int* in_sizes, int n_in,
                              void* d_out, int out_size, void* d_ws, size_t ws_size,
                              hipStream_t stream) {
    const int B = 16384, C = 768;
    const float* x     = (const float*)d_in[0];
    const float* y     = (const float*)d_in[1];
    const float* y2x   = (const float*)d_in[2];
    const float* x2y   = (const float*)d_in[3];
    const float* Wqkv  = (const float*)d_in[4];
    const float* Wpb   = (const float*)d_in[5];
    const float* bpb   = (const float*)d_in[6];
    const float* Wproj = (const float*)d_in[7];
    const float* bproj = (const float*)d_in[8];
    const float* ln_g  = (const float*)d_in[9];
    const float* ln_b  = (const float*)d_in[10];

    // padded block counts
    const int KBP_C   = C / 32;          // 24 (even)
    const int NBP_QKV = (3 * C) / 16;    // 144
    const int NBP_PROJ = C / 16;         // 48
    const int KBP_PB  = 4;               // K=96 -> pad to 128 (even)
    const int NBP_PB  = 8;               // N=96 -> pad to 128 cols

    char* w = (char*)d_ws;
    float*          lnres  = (float*)w;          w += (size_t)4 * B * C * 4;      // LN(x,y,y2x,x2y) f32
    unsigned short* abuf   = (unsigned short*)w; w += (size_t)2 * B * C * 2;      // LN(x,y) bf16
    float*          qkvbuf = (float*)w;          w += (size_t)2 * B * 3 * C * 4;  // qkv f32 (later aliased as fpre)
    unsigned short* gbuf   = (unsigned short*)w; w += (size_t)4 * B * 8 * 128 * 2; // attn*v bf16 (K-padded), then h bf16
    unsigned short* wqkvp  = (unsigned short*)w; w += (size_t)KBP_C  * NBP_QKV  * 512 * 2;
    unsigned short* wprojp = (unsigned short*)w; w += (size_t)KBP_C  * NBP_PROJ * 512 * 2;
    unsigned short* wpbp   = (unsigned short*)w; w += (size_t)KBP_PB * NBP_PB   * 512 * 2;
    float* fpre = qkvbuf;                        // alias: qkv dead after attn_kernel
    float* hout = (float*)d_out;                 // d_out temporarily holds h (f32)

    // 1) pack weights to bf16 fragment layout (zero-padded)
    {
        long tq = (long)KBP_C  * NBP_QKV  * 512;
        long tp = (long)KBP_C  * NBP_PROJ * 512;
        long tb = (long)KBP_PB * NBP_PB   * 512;
        pack_weight_bf16<<<dim3((unsigned)((tq + 255) / 256)), 256, 0, stream>>>(Wqkv,  wqkvp,  C,  3 * C, KBP_C,  NBP_QKV);
        pack_weight_bf16<<<dim3((unsigned)((tp + 255) / 256)), 256, 0, stream>>>(Wproj, wprojp, C,  C,     KBP_C,  NBP_PROJ);
        pack_weight_bf16<<<dim3((unsigned)((tb + 255) / 256)), 256, 0, stream>>>(Wpb,   wpbp,   96, 96,    KBP_PB, NBP_PB);
    }

    // 2) LayerNorm inputs (x,y also emitted as bf16 GEMM operands)
    ln_kernel<<<B, 256, 0, stream>>>(x,   nullptr, ln_g, ln_b, lnres,                     abuf);
    ln_kernel<<<B, 256, 0, stream>>>(y,   nullptr, ln_g, ln_b, lnres + (size_t)B * C,     abuf + (size_t)B * C);
    ln_kernel<<<B, 256, 0, stream>>>(y2x, nullptr, ln_g, ln_b, lnres + (size_t)2 * B * C, nullptr);
    ln_kernel<<<B, 256, 0, stream>>>(x2y, nullptr, ln_g, ln_b, lnres + (size_t)3 * B * C, nullptr);

    // 3) QKV GEMM: [2B,768] @ [768,2304]
    gemm_bf16<<<dim3((2 * B) / 128, (3 * C) / 64), 256, 0, stream>>>(
        abuf, wqkvp, nullptr, nullptr, qkvbuf, 2 * B, 3 * C, C);

    // 4) elementwise softmax(q*k)*v for the 4 combos (writes K-padded rows)
    attn_kernel<<<4 * B, 256, 0, stream>>>(qkvbuf, gbuf, B);

    // 5) per-head Wpb GEMM: [4B*8, 128pad] @ [128pad, 128pad] + bpb -> [., 96]
    gemm_bf16<<<dim3((4 * B * 8) / 128, 2), 256, 0, stream>>>(
        gbuf, wpbp, bpb, nullptr, fpre, 4 * B * 8, 96, 128);

    // 6) residual add + LN -> h (f32 into d_out, bf16 into gbuf)
    ln_kernel<<<4 * B, 256, 0, stream>>>(fpre, lnres, ln_g, ln_b, hout, gbuf);

    // 7) projection GEMM: [4B,768] @ [768,768] + bproj + h -> d_out
    gemm_bf16<<<dim3((4 * B) / 128, C / 64), 256, 0, stream>>>(
        gbuf, wprojp, bproj, hout, (float*)d_out, 4 * B, C, C);
}
